// Net_21689584845426
// MI455X (gfx1250) — matrix operand order repaired and checked by reading, not compiled, not run on hardware
//
#include <hip/hip_runtime.h>
#include <stddef.h>

// ---------------- problem constants ----------------
#define NN   50000      // nodes
#define RR   50         // relations
#define BB   30         // bases
#define DIN  64         // input feature dim (all 3 layers)
#define EE   1000000    // edges
#define NR   (NN * RR)              // 2.5M segments
#define KAGG (RR * DIN)             // 3200 (aggregated part of K)
#define KTOT (KAGG + DIN)           // 3264 = full K (agg + root block); A row stride
#define KSTEPS (KTOT / 32)          // 102 WMMA K-steps of 32
#define NTILES (NN / 16)            // 3125 node tiles

typedef __bf16 v16bf __attribute__((ext_vector_type(16)));
typedef __bf16 v8bf  __attribute__((ext_vector_type(8)));
typedef float  v8f   __attribute__((ext_vector_type(8)));

union AFrag { v16bf v; v8bf h[2]; };

// ---------------- stage 0: edge counts -> inverse counts (shared by all layers)
__global__ void count_kernel(const int* __restrict__ ei, const int* __restrict__ et,
                             unsigned int* __restrict__ cnt) {
    int e = blockIdx.x * blockDim.x + threadIdx.x;
    if (e >= EE) return;
    int dst = ei[EE + e];
    int rel = et[e];
    atomicAdd(&cnt[(size_t)dst * RR + rel], 1u);
}

__global__ void inv_kernel(const unsigned int* __restrict__ cnt, float* __restrict__ inv) {
    int i = blockIdx.x * blockDim.x + threadIdx.x;
    if (i >= NR) return;
    unsigned int c = cnt[i];
    inv[i] = c ? (1.0f / (float)c) : 0.0f;
}

__global__ void x2bf_kernel(const float* __restrict__ x, __bf16* __restrict__ xbf) {
    int i = blockIdx.x * blockDim.x + threadIdx.x;
    if (i >= NN * DIN) return;
    xbf[i] = (__bf16)x[i];
}

// Copy x rows into the last 64 columns of agg -> A becomes one uniform
// [N, 3264] bf16 matrix (root block folded in); 16B per thread.
__global__ void tailcopy_kernel(const __bf16* __restrict__ xbf, __bf16* __restrict__ agg) {
    int gid = blockIdx.x * blockDim.x + threadIdx.x;
    if (gid >= NN * 8) return;
    int row = gid >> 3;
    int q   = gid & 7;
    *(v8bf*)(agg + (size_t)row * KTOT + KAGG + q * 8) =
        *(const v8bf*)(xbf + (size_t)row * DIN + q * 8);
}

// ---------------- stage 1: edge scatter -> bf16 mean buffer -----------------
// One wave per edge; lane handles 2 features via global_atomic_pk_add_bf16.
// Mean scaling 1/cnt is folded in per-term (linear). Segment (dst,rel) lives
// at agg[dst, rel*64 .. rel*64+63] (row stride KTOT).
__global__ void scatter_kernel(const int* __restrict__ ei, const int* __restrict__ et,
                               const float* __restrict__ inv,
                               const __bf16* __restrict__ xbf,
                               __bf16* __restrict__ agg) {
    int gid  = blockIdx.x * blockDim.x + threadIdx.x;
    int e    = gid >> 5;
    int lane = gid & 31;
    if (e >= EE) return;
    int src = ei[e];
    int dst = ei[EE + e];
    int rel = et[e];
    float s = inv[(size_t)dst * RR + rel];
    float f0 = (float)xbf[(size_t)src * DIN + 2 * lane]     * s;
    float f1 = (float)xbf[(size_t)src * DIN + 2 * lane + 1] * s;
    union { __bf16 h[2]; unsigned int u; } p;
    p.h[0] = (__bf16)f0;
    p.h[1] = (__bf16)f1;
    unsigned long long addr =
        (unsigned long long)(agg + (size_t)dst * KTOT + rel * DIN + 2 * lane);
    asm volatile("global_atomic_pk_add_bf16 %0, %1, off"
                 :: "v"(addr), "v"(p.u) : "memory");
}

// ---------------- stage 2: build Wbig in B-fragment-swizzled order ----------
// wfrag[((ct*KSTEPS + ks)*32 + lane)*16 + j] : lane gives N=lane%16,
// K = ks*32 + (lane<16 ? 0 : 8) + (j&7) + (j>=8 ? 16 : 0)   (ISA 7.12.2 dense 16-bit B layout).
__global__ void wbig_kernel(const float* __restrict__ bases, const float* __restrict__ comp,
                            const float* __restrict__ root, int dout, int nct,
                            __bf16* __restrict__ wfrag) {
    int idx = blockIdx.x * blockDim.x + threadIdx.x;
    int total = nct * KSTEPS * 32 * 16;
    if (idx >= total) return;
    int j    = idx & 15;
    int lane = (idx >> 4) & 31;
    int ks   = (idx >> 9) % KSTEPS;
    int ct   = idx / (KSTEPS * 512);
    int col  = ct * 16 + (lane & 15);
    int kg   = ks * 32 + ((lane < 16) ? 0 : 8) + (j & 7) + ((j & 8) << 1);
    float val = 0.0f;
    if (col < dout) {
        if (kg < KAGG) {
            int r = kg >> 6;
            int i = kg & 63;
            float acc = 0.0f;
            for (int b = 0; b < BB; ++b)
                acc += comp[r * BB + b] * bases[((size_t)b * DIN + i) * dout + col];
            val = acc;
        } else {
            val = root[(size_t)(kg - KAGG) * dout + col];
        }
    }
    wfrag[idx] = (__bf16)val;
}

// ---------------- stage 3: fused RGCN layer as GEMM with WMMA ---------------
// One 16-node tile per WAVE; the wave owns all NCOLT 16-column tiles.
// A fragment = two contiguous b128 loads per lane straight from global
// (ISA 7.12.2: lane's 16 A elements are rows m=lane%16, K in two 8-element
//  runs at kadd and 16+kadd, kadd = (lane<16)?0:8). No LDS, no barriers.
// Software-pipelined depth 2 (ping-pong A/B register sets) so the loads for
// step k+1 are in flight while the NCOLT WMMAs of step k execute.
template <int NCOLT, bool FINAL>
__global__ __launch_bounds__(256)
void gemm_kernel(const __bf16* __restrict__ agg,     // [N, KTOT] bf16
                 const __bf16* __restrict__ wfrag,   // [NCOLT][KSTEPS][32][16]
                 const float*  __restrict__ bias,
                 __bf16* __restrict__ xnext,
                 float*  __restrict__ out) {
    const int lane = threadIdx.x & 31;
    const int wv   = threadIdx.x >> 5;
    const int tile = blockIdx.x * 8 + wv;
    if (tile >= NTILES) return;
    const int n0   = tile * 16;
    const int m    = lane & 15;
    const int kadd = (lane & 16) >> 1;          // 0 or 8

    v8f c[NCOLT];
#pragma unroll
    for (int t = 0; t < NCOLT; ++t) c[t] = (v8f){};

    const __bf16* abase = agg + (size_t)(n0 + m) * KTOT + kadd;  // 16B aligned
    const __bf16* wp    = wfrag + ((size_t)lane << 4);           // 32B aligned

    AFrag a0, a1;
    v16bf b0[NCOLT], b1[NCOLT];

    auto loadfrag = [&](int ks, AFrag& a, v16bf* b) {
        const __bf16* ap = abase + (size_t)ks * 32;
        a.h[0] = *(const v8bf*)(ap);
        a.h[1] = *(const v8bf*)(ap + 16);
#pragma unroll
        for (int t = 0; t < NCOLT; ++t)
            b[t] = *(const v16bf*)(wp + (size_t)t * (KSTEPS * 512) + (size_t)ks * 512);
    };
    auto mma = [&](const AFrag& a, const v16bf* b) {
#pragma unroll
        for (int t = 0; t < NCOLT; ++t)
            c[t] = __builtin_amdgcn_wmma_f32_16x16x32_bf16(false, a.v, false, b[t],
                                                           (short)0, c[t], false, false);
    };

    loadfrag(0, a0, b0);
    for (int ks = 0; ks < KSTEPS - 2; ks += 2) {
        loadfrag(ks + 1, a1, b1);
        mma(a0, b0);
        loadfrag(ks + 2, a0, b0);
        mma(a1, b1);
    }
    loadfrag(KSTEPS - 1, a1, b1);
    mma(a0, b0);      // frag KSTEPS-2
    mma(a1, b1);      // frag KSTEPS-1

    // epilogue: C/D layout — VGPR i holds M = i + (lane<16 ? 0 : 8), N = lane%16
    const int mbase = (lane & 16) ? 8 : 0;
#pragma unroll
    for (int t = 0; t < NCOLT; ++t) {
        const int col = t * 16 + m;
#pragma unroll
        for (int i = 0; i < 8; ++i) {
            int n = n0 + mbase + i;
            if (FINAL) {
                if (col < 8) out[(size_t)n * 8 + col] = c[t][i] + bias[col];
            } else {
                float v = c[t][i] + bias[col];
                xnext[(size_t)n * DIN + col] = (__bf16)(v > 0.0f ? v : 0.0f);
            }
        }
    }
}

// ---------------- stage 4: log_softmax over 8 classes -----------------------
__global__ void logsoftmax_kernel(float* __restrict__ out) {
    int n = blockIdx.x * blockDim.x + threadIdx.x;
    if (n >= NN) return;
    float v[8], m = -1e30f;
#pragma unroll
    for (int i = 0; i < 8; ++i) { v[i] = out[(size_t)n * 8 + i]; m = fmaxf(m, v[i]); }
    float s = 0.0f;
#pragma unroll
    for (int i = 0; i < 8; ++i) s += expf(v[i] - m);
    float ls = logf(s);
#pragma unroll
    for (int i = 0; i < 8; ++i) out[(size_t)n * 8 + i] = v[i] - m - ls;
}

// ---------------- launch ----------------------------------------------------
extern "C" void kernel_launch(void* const* d_in, const int* in_sizes, int n_in,
                              void* d_out, int out_size, void* d_ws, size_t ws_size,
                              hipStream_t stream) {
    (void)in_sizes; (void)n_in; (void)out_size; (void)ws_size;
    const float* x  = (const float*)d_in[0];
    const int*   ei = (const int*)d_in[1];
    const int*   et = (const int*)d_in[2];
    const float* bases[3] = {(const float*)d_in[3], (const float*)d_in[7],  (const float*)d_in[11]};
    const float* comp[3]  = {(const float*)d_in[4], (const float*)d_in[8],  (const float*)d_in[12]};
    const float* root[3]  = {(const float*)d_in[5], (const float*)d_in[9],  (const float*)d_in[13]};
    const float* bias[3]  = {(const float*)d_in[6], (const float*)d_in[10], (const float*)d_in[14]};

    char*  ws  = (char*)d_ws;
    size_t off = 0;
    auto alloc = [&](size_t bytes) -> void* {
        void* p = ws + off;
        off = (off + bytes + 255) & ~(size_t)255;
        return p;
    };
    unsigned int* cnt  = (unsigned int*)alloc((size_t)NR * 4);
    float*        inv  = (float*)alloc((size_t)NR * 4);
    __bf16*       xbf0 = (__bf16*)alloc((size_t)NN * DIN * 2);
    __bf16*       xbf1 = (__bf16*)alloc((size_t)NN * DIN * 2);
    __bf16*       wfrg = (__bf16*)alloc((size_t)4 * KSTEPS * 32 * 16 * 2);
    __bf16*       agg  = (__bf16*)alloc((size_t)NN * KTOT * 2);   // 326 MB bf16

    hipMemsetAsync(cnt, 0, (size_t)NR * 4, stream);
    count_kernel<<<(EE + 255) / 256, 256, 0, stream>>>(ei, et, cnt);
    inv_kernel<<<(NR + 255) / 256, 256, 0, stream>>>(cnt, inv);
    x2bf_kernel<<<(NN * DIN + 255) / 256, 256, 0, stream>>>(x, xbf0);

    __bf16* xc = xbf0;
    __bf16* xn = xbf1;
    const int gemm_blocks = (NTILES + 7) / 8;   // 8 waves (node tiles) per block
    for (int l = 0; l < 3; ++l) {
        hipMemsetAsync(agg, 0, (size_t)NN * KTOT * 2, stream);
        tailcopy_kernel<<<(NN * 8 + 255) / 256, 256, 0, stream>>>(xc, agg);
        scatter_kernel<<<(EE * 32) / 256, 256, 0, stream>>>(ei, et, inv, xc, agg);
        int dout = (l == 2) ? 8 : 64;
        int nct  = (l == 2) ? 1 : 4;
        int wtot = nct * KSTEPS * 32 * 16;
        wbig_kernel<<<(wtot + 255) / 256, 256, 0, stream>>>(bases[l], comp[l], root[l],
                                                            dout, nct, wfrg);
        if (l < 2) {
            gemm_kernel<4, false><<<gemm_blocks, 256, 0, stream>>>(agg, wfrg, bias[l],
                                                                   xn, nullptr);
        } else {
            gemm_kernel<1, true><<<gemm_blocks, 256, 0, stream>>>(agg, wfrg, bias[l],
                                                                  nullptr, (float*)d_out);
        }
        __bf16* tmp = xc; xc = xn; xn = tmp;
    }
    logsoftmax_kernel<<<(NN + 255) / 256, 256, 0, stream>>>((float*)d_out);
}
